// DilatedAttention_17927193493964
// MI455X (gfx1250) — matrix operand order repaired and checked by reading, not compiled, not run on hardware
//
#include <hip/hip_runtime.h>

// ---------------------------------------------------------------------------
// Types for CDNA5 WMMA
// ---------------------------------------------------------------------------
typedef __attribute__((ext_vector_type(16))) __bf16 v16bf;
typedef __attribute__((ext_vector_type(8)))  float  v8f;

__device__ __forceinline__ unsigned short f32_to_bf16(float f) {
    unsigned u = __float_as_uint(f);
    unsigned r = u + 0x7FFFu + ((u >> 16) & 1u);   // round-to-nearest-even
    return (unsigned short)(r >> 16);
}
__device__ __forceinline__ unsigned pack_bf16x2(float lo, float hi) {
    return (unsigned)f32_to_bf16(lo) | ((unsigned)f32_to_bf16(hi) << 16);
}

// 32 bf16 payload + pad -> 40 ushorts (80B = 20 dwords). 16B-aligned rows for
// ds_store_b128; 20*m mod 64 distinct for m in [0,16) -> conflict-light reads.
#define LDS_STRIDE 40

// Output modes for the GEMM epilogue
#define OUT_F32   0   // C float [M,N], optional bias / accumulate, alpha
#define OUT_BF16  1   // C bf16 [M,N], optional bias
#define OUT_BF16T 2   // C bf16 [N,M] (B-layout for a later GEMM), optional bias

// ---------------------------------------------------------------------------
// Tiled GEMM on pre-converted bf16 operands:
//   A: bf16 [M,K] row-major (lda).  B: bf16 [N,K] row-major (ldb).
//   C = alpha * A x B^T(layout) [+ bias] [+ C]
// Block tile 128x128, 8 wave32s; each wave 32x64 via 2x4 wmma f32_16x16x32_bf16.
// Requires M%128==0, N%128==0, K%32==0 (true for all launches below).
// ---------------------------------------------------------------------------
template <int OUTMODE, bool HASBIAS, bool ACC>
__global__ __launch_bounds__(256)
void gemm_bf16_kernel(const unsigned short* __restrict__ A, int lda, long long strideA,
                      const unsigned short* __restrict__ B, int ldb, long long strideB,
                      void* __restrict__ Cv, int ldc, long long strideC,
                      const float* __restrict__ bias,
                      int M, int N, int K, float alpha)
{
    __shared__ unsigned short As[128 * LDS_STRIDE];
    __shared__ unsigned short Bs[128 * LDS_STRIDE];

    const int tid = threadIdx.x;
    const int bm  = blockIdx.y * 128;
    const int bn  = blockIdx.x * 128;
    const int bz  = blockIdx.z;

    A += (size_t)bz * (size_t)strideA;
    B += (size_t)bz * (size_t)strideB;

    const int wave  = tid >> 5;
    const int lane  = tid & 31;
    const int wm    = wave >> 1;        // 0..3  (row block of 32)
    const int wn    = wave & 1;         // 0..1  (col block of 64)
    const int lhalf = lane >> 4;        // 0/1
    const int l16   = lane & 15;

    v8f zero = {0.f,0.f,0.f,0.f,0.f,0.f,0.f,0.f};
    v8f acc[2][4];
    #pragma unroll
    for (int mt = 0; mt < 2; ++mt)
        #pragma unroll
        for (int nt = 0; nt < 4; ++nt) acc[mt][nt] = zero;

    // Staging: 256 threads, 128x32 bf16 tile, 8 contiguous bf16 (16B) each.
    const int sr = tid >> 1;            // 0..127
    const int sc = (tid & 1) * 16;      // 0 or 16

    union Frag { unsigned u[8]; v16bf v; };

    for (int k0 = 0; k0 < K; k0 += 32) {
        // Branch-free prefetch distance: next K-chunk, clamped in-bounds.
        const int pfd = (k0 + 32 < K) ? 32 : 0;
        {   // A tile
            const unsigned short* ap = A + (size_t)(bm + sr) * (size_t)lda + (size_t)(k0 + sc);
            uint4 wa = *(const uint4*)ap;
            __builtin_prefetch(ap + pfd, 0, 3);
            *(uint4*)&As[sr * LDS_STRIDE + sc] = wa;
        }
        {   // B tile (already [N,K]: contiguous along k)
            const unsigned short* bp = B + (size_t)(bn + sr) * (size_t)ldb + (size_t)(k0 + sc);
            uint4 wb = *(const uint4*)bp;
            __builtin_prefetch(bp + pfd, 0, 3);
            *(uint4*)&Bs[sr * LDS_STRIDE + sc] = wb;
        }
        __syncthreads();

        // Build fragments: element s holds k = 16*lhalf + s (contiguous order), same for A and B
        Frag fa[2], fb[4];
        #pragma unroll
        for (int v = 0; v < 8; ++v) {
            const int ka = (v << 1) + (lhalf << 4);                         // A 16x32
            const int kb = (v << 1) + (lhalf << 4);                         // B 32x16
            #pragma unroll
            for (int mt = 0; mt < 2; ++mt)
                fa[mt].u[v] = *(const unsigned*)&As[(wm * 32 + mt * 16 + l16) * LDS_STRIDE + ka];
            #pragma unroll
            for (int nt = 0; nt < 4; ++nt)
                fb[nt].u[v] = *(const unsigned*)&Bs[(wn * 64 + nt * 16 + l16) * LDS_STRIDE + kb];
        }

        #pragma unroll
        for (int mt = 0; mt < 2; ++mt)
            #pragma unroll
            for (int nt = 0; nt < 4; ++nt)
                acc[mt][nt] = __builtin_amdgcn_wmma_f32_16x16x32_bf16(
                    false, fa[mt].v, false, fb[nt].v,
                    (short)0, acc[mt][nt], false, false);

        __syncthreads();
    }

    // ---- epilogue (branch-free per template) ----
    #pragma unroll
    for (int mt = 0; mt < 2; ++mt) {
        #pragma unroll
        for (int nt = 0; nt < 4; ++nt) {
            const int gn     = bn + wn * 64 + nt * 16 + l16;
            const int gmBase = bm + wm * 32 + mt * 16 + lhalf * 8;
            float bval = 0.0f;
            if (HASBIAS) bval = bias[gn];

            if (OUTMODE == OUT_F32) {
                float* Cf = (float*)Cv + (size_t)bz * (size_t)strideC;
                #pragma unroll
                for (int v = 0; v < 8; ++v) {
                    float* cp = Cf + (size_t)(gmBase + v) * (size_t)ldc + (size_t)gn;
                    float val = alpha * acc[mt][nt][v] + bval;
                    if (ACC) val += *cp;
                    *cp = val;
                }
            } else if (OUTMODE == OUT_BF16) {
                unsigned short* Ch = (unsigned short*)Cv + (size_t)bz * (size_t)strideC;
                #pragma unroll
                for (int v = 0; v < 8; ++v)
                    Ch[(size_t)(gmBase + v) * (size_t)ldc + (size_t)gn] =
                        f32_to_bf16(acc[mt][nt][v] + bval);
            } else { // OUT_BF16T: T[n][m], 8 contiguous bf16 -> one 16B store
                unsigned short* Ch = (unsigned short*)Cv + (size_t)bz * (size_t)strideC;
                uint4 w;
                w.x = pack_bf16x2(acc[mt][nt][0] + bval, acc[mt][nt][1] + bval);
                w.y = pack_bf16x2(acc[mt][nt][2] + bval, acc[mt][nt][3] + bval);
                w.z = pack_bf16x2(acc[mt][nt][4] + bval, acc[mt][nt][5] + bval);
                w.w = pack_bf16x2(acc[mt][nt][6] + bval, acc[mt][nt][7] + bval);
                *(uint4*)&Ch[(size_t)gn * (size_t)ldc + (size_t)gmBase] = w;
            }
        }
    }
}

// ---------------------------------------------------------------------------
// Masked row softmax, S==2048, 256 threads, 8 elements/thread in registers.
// mask: (j % rate == 0) && (j <= i). Masked entries are exactly 0 (matches
// exp(-1e9 - m) underflow). Writes bf16 P (GEMM A operand) and accumulates
// 0.25*p (fp32) into avg_attention.
// ---------------------------------------------------------------------------
__global__ __launch_bounds__(256)
void softmax_mask_kernel(const float* __restrict__ scores,
                         unsigned short* __restrict__ P, float* __restrict__ avg,
                         int S, int rateMask, int first)
{
    __shared__ float red[256];
    const int row = blockIdx.x;            // b*S + i
    const int i   = row % S;
    const int tid = threadIdx.x;
    const float* srow = scores + (size_t)row * (size_t)S;
    unsigned short* prow = P + (size_t)row * (size_t)S;
    float* arow = avg + (size_t)row * (size_t)S;

    float sv[8];
    float lmax = -3.0e38f;
    #pragma unroll
    for (int t = 0; t < 8; ++t) {
        const int j = t * 256 + tid;
        const float s = srow[j];
        const bool allowed = ((j & rateMask) == 0) && (j <= i);
        sv[t] = allowed ? s : -3.0e38f;
        if (allowed) lmax = fmaxf(lmax, s);
    }
    red[tid] = lmax; __syncthreads();
    for (int s = 128; s > 0; s >>= 1) {
        if (tid < s) red[tid] = fmaxf(red[tid], red[tid + s]);
        __syncthreads();
    }
    const float m = red[0];
    __syncthreads();

    float ev[8];
    float lsum = 0.0f;
    #pragma unroll
    for (int t = 0; t < 8; ++t) {
        ev[t] = (sv[t] > -1.0e37f) ? __expf(sv[t] - m) : 0.0f;
        lsum += ev[t];
    }
    red[tid] = lsum; __syncthreads();
    for (int s = 128; s > 0; s >>= 1) {
        if (tid < s) red[tid] += red[tid + s];
        __syncthreads();
    }
    const float inv = 1.0f / red[0];

    #pragma unroll
    for (int t = 0; t < 8; ++t) {
        const int j = t * 256 + tid;
        const float p = ev[t] * inv;
        prow[j] = f32_to_bf16(p);
        float a = 0.25f * p;
        if (!first) a += arow[j];
        arow[j] = a;
    }
}

// ---------------------------------------------------------------------------
// One-time conversion kernels
// ---------------------------------------------------------------------------
__global__ __launch_bounds__(256)
void cvt_bf16_kernel(const float* __restrict__ src, unsigned short* __restrict__ dst,
                     long long n)
{
    long long idx = (long long)blockIdx.x * 256 + threadIdx.x;
    const long long stride = (long long)gridDim.x * 256;
    for (; idx < n; idx += stride) dst[idx] = f32_to_bf16(src[idx]);
}

// dst[c][r] = bf16(src[r][c]);  src: [rows, cols] f32 -> dst: [cols, rows] bf16
__global__ __launch_bounds__(256)
void cvtT_bf16_kernel(const float* __restrict__ src, unsigned short* __restrict__ dst,
                      int rows, int cols)
{
    long long idx = (long long)blockIdx.x * 256 + threadIdx.x;
    const long long n = (long long)rows * cols;
    if (idx < n) {
        const int r = (int)(idx / cols);
        const int c = (int)(idx % cols);
        dst[(size_t)c * rows + r] = f32_to_bf16(src[idx]);
    }
}

// beff[n] = bo[n] + sum_t br_flat[t] * Wo[t*512 + n]   (t over R*D = 2048)
__global__ __launch_bounds__(256)
void bias_eff_kernel(const float* __restrict__ br, const float* __restrict__ Wo,
                     const float* __restrict__ bo, float* __restrict__ beff)
{
    const int n = blockIdx.x * blockDim.x + threadIdx.x;
    if (n < 512) {
        float s = bo[n];
        for (int t = 0; t < 2048; ++t) s += br[t] * Wo[(size_t)t * 512 + n];
        beff[n] = s;
    }
}

// ---------------------------------------------------------------------------
// Host-side orchestration
// ---------------------------------------------------------------------------
extern "C" void kernel_launch(void* const* d_in, const int* in_sizes, int n_in,
                              void* d_out, int out_size, void* d_ws, size_t ws_size,
                              hipStream_t stream)
{
    (void)in_sizes; (void)n_in; (void)out_size; (void)ws_size;

    const float* x  = (const float*)d_in[0];
    const float* Wq = (const float*)d_in[1];
    const float* bq = (const float*)d_in[2];
    const float* Wk = (const float*)d_in[3];
    const float* bk = (const float*)d_in[4];
    const float* Wv = (const float*)d_in[5];
    const float* bv = (const float*)d_in[6];
    const float* Wr = (const float*)d_in[7];
    const float* br = (const float*)d_in[8];
    const float* Wo = (const float*)d_in[9];
    const float* bo = (const float*)d_in[10];

    const int Bb = 2, S = 2048, D = 512;
    const long long SD = (long long)S * D;          // 1,048,576
    const long long SS = (long long)S * S;          // 4,194,304
    const int BS = Bb * S;                          // 4096
    const long long DD = (long long)D * D;

    float* out = (float*)d_out;                     // [B,S,D]
    float* avg = out + (size_t)Bb * S * D;          // [B,S,S]

    // ---- workspace carve-up (256B-aligned chunks) ----
    char* base = (char*)d_ws;
    size_t off = 0;
    auto alloc = [&](size_t bytes) -> void* {
        void* p = base + off;
        off += (bytes + 255) & ~(size_t)255;
        return p;
    };
    float*          Sc    = (float*)         alloc((size_t)Bb * S * S * 4);   // scores f32
    float*          be    = (float*)         alloc((size_t)D * 4);            // eff bias
    unsigned short* xbf   = (unsigned short*)alloc((size_t)BS * D * 2);       // x bf16 [4096,512]
    unsigned short* WqT   = (unsigned short*)alloc((size_t)DD * 2);           // [512,512] (B-layout)
    unsigned short* WkT   = (unsigned short*)alloc((size_t)DD * 2);
    unsigned short* WvT   = (unsigned short*)alloc((size_t)DD * 2);
    unsigned short* WoT   = (unsigned short*)alloc((size_t)4 * DD * 2);       // [512,2048]
    unsigned short* Wrbf  = (unsigned short*)alloc((size_t)4 * DD * 2);       // [4,512,512] (A)
    unsigned short* Qbf   = (unsigned short*)alloc((size_t)BS * D * 2);       // [2,2048,512] (A)
    unsigned short* Kbf   = (unsigned short*)alloc((size_t)BS * D * 2);       // [2,2048,512] (B-layout!)
    unsigned short* VT    = (unsigned short*)alloc((size_t)BS * D * 2);       // [512,4096] (B-layout)
    unsigned short* Pbf   = (unsigned short*)alloc((size_t)Bb * S * S * 2);   // attn bf16 (A)
    unsigned short* ctxbf = (unsigned short*)alloc((size_t)BS * D * 2);       // [4096,512] (A)
    unsigned short* WeffT = (unsigned short*)alloc((size_t)4 * DD * 2);       // [4][512,512] (B-layout)

    const dim3 blk(256);
    const float inv_sqrt_d = 0.044194173824159216f;    // 1/sqrt(512)

    // ---- one-time conversions ----
    cvt_bf16_kernel<<<dim3(1024), blk, 0, stream>>>(x, xbf, (long long)BS * D);
    cvt_bf16_kernel<<<dim3(1024), blk, 0, stream>>>(Wr, Wrbf, 4 * DD);
    cvtT_bf16_kernel<<<dim3((unsigned)(DD / 256)), blk, 0, stream>>>(Wq, WqT, D, D);
    cvtT_bf16_kernel<<<dim3((unsigned)(DD / 256)), blk, 0, stream>>>(Wk, WkT, D, D);
    cvtT_bf16_kernel<<<dim3((unsigned)(DD / 256)), blk, 0, stream>>>(Wv, WvT, D, D);
    cvtT_bf16_kernel<<<dim3((unsigned)(4 * DD / 256)), blk, 0, stream>>>(Wo, WoT, 4 * D, D);
    bias_eff_kernel<<<dim3(2), blk, 0, stream>>>(br, Wo, bo, be);

    // ---- WeffT[r] = (Wr[r] @ Wo_r)^T, bf16 B-layout, via transposed epilogue ----
    // A = Wrbf[r] [512,512]; B = WoT + r*512 (ldb=2048) => B[n][k] = Wo[r*512+k][n]
    gemm_bf16_kernel<OUT_BF16T, false, false><<<dim3(4, 4, 4), blk, 0, stream>>>(
        Wrbf, D, DD, WoT + (size_t)0, 4 * D, (long long)D,
        WeffT, D, DD, nullptr, D, D, D, 1.0f);
    // note: strideB = 512 elements walks WoT columns per r; strideC = DD per r.

    // ---- Q/K/V projections ----
    gemm_bf16_kernel<OUT_BF16, true, false><<<dim3(4, 32, 1), blk, 0, stream>>>(
        xbf, D, 0, WqT, D, 0, Qbf, D, 0, bq, BS, D, D, 1.0f);
    gemm_bf16_kernel<OUT_BF16, true, false><<<dim3(4, 32, 1), blk, 0, stream>>>(
        xbf, D, 0, WkT, D, 0, Kbf, D, 0, bk, BS, D, D, 1.0f);
    // V written transposed: VT[n][m], m = b*2048 + s, ldc = 4096
    gemm_bf16_kernel<OUT_BF16T, true, false><<<dim3(4, 32, 1), blk, 0, stream>>>(
        xbf, D, 0, WvT, D, 0, VT, BS, 0, bv, BS, D, D, 1.0f);

    // ---- scores[b] = Q[b] @ K[b]^T / sqrt(D) ----
    gemm_bf16_kernel<OUT_F32, false, false><<<dim3(16, 16, 2), blk, 0, stream>>>(
        Qbf, D, SD, Kbf, D, SD, Sc, S, SS, nullptr, S, S, D, inv_sqrt_d);

    // ---- per-rate: softmax (+avg) -> ctx = P@V -> out (+)= ctx @ Weff[r] ----
    for (int r = 0; r < 4; ++r) {
        const int rate = 1 << r;
        softmax_mask_kernel<<<dim3(Bb * S), blk, 0, stream>>>(
            Sc, Pbf, avg, S, rate - 1, (r == 0) ? 1 : 0);

        // ctx[b] = P[b] @ V[b]; B = VT (ldb=4096), batch b offset = b*2048 columns
        gemm_bf16_kernel<OUT_BF16, false, false><<<dim3(4, 16, 2), blk, 0, stream>>>(
            Pbf, S, SS, VT, BS, (long long)S, ctxbf, D, SD,
            nullptr, S, D, S, 1.0f);

        if (r == 0) {
            gemm_bf16_kernel<OUT_F32, true, false><<<dim3(4, 32, 1), blk, 0, stream>>>(
                ctxbf, D, 0, WeffT + (size_t)r * DD, D, 0, out, D, 0,
                be, BS, D, D, 1.0f);
        } else {
            gemm_bf16_kernel<OUT_F32, false, true><<<dim3(4, 32, 1), blk, 0, stream>>>(
                ctxbf, D, 0, WeffT + (size_t)r * DD, D, 0, out, D, 0,
                nullptr, BS, D, D, 1.0f);
        }
    }
}
